// TypeHierarchyEncoder_39874476376561
// MI455X (gfx1250) — compile-verified, matrix operand order checked by reading, not run on hardware
//
#include <hip/hip_runtime.h>
#include <hip/hip_bf16.h>

typedef __attribute__((ext_vector_type(16))) __bf16 v16bf;
typedef __attribute__((ext_vector_type(8)))  float  v8f;

// ---------- helpers ----------
__device__ __forceinline__ unsigned short f2bf(float f) {
    unsigned int u = __float_as_uint(f);
    unsigned int r = u + 0x7FFFu + ((u >> 16) & 1u);   // round-to-nearest-even
    return (unsigned short)(r >> 16);
}

// K index inside a 32-wide slab for element j of a 16-bit WMMA A/B fragment,
// per CDNA5 ISA 7.12.2 (16-bit A-matrix 16x32 layout; B mirrors it over N).
__device__ __forceinline__ int kpat(int j, int lane) {
    int vg   = j >> 1;
    int base = (vg < 4) ? (vg * 2) : (16 + (vg - 4) * 2);
    return base + (j & 1) + ((lane >> 4) << 3);
}

// ---------- utility kernels ----------
__global__ void zero_kernel(float* __restrict__ p, int n) {
    int i = blockIdx.x * blockDim.x + threadIdx.x;
    int stride = gridDim.x * blockDim.x;
    for (; i < n; i += stride) p[i] = 0.0f;
}

__global__ void count_kernel(const int* __restrict__ dst, float* __restrict__ cnt, int E) {
    int e = blockIdx.x * blockDim.x + threadIdx.x;
    if (e < E) unsafeAtomicAdd(&cnt[dst[e]], 1.0f);
}

// one thread per (edge, 4-feature chunk): float4 gather + 4 HW f32 atomics
template<int FIN>
__global__ void scatter_kernel(const float* __restrict__ xin,
                               const int* __restrict__ src,
                               const int* __restrict__ dst,
                               float* __restrict__ agg,
                               int E) {
    constexpr int CH = FIN / 4;
    long long t = (long long)blockIdx.x * blockDim.x + threadIdx.x;
    long long nwork = (long long)E * CH;
    if (t >= nwork) return;
    int e = (int)(t / CH);
    int c = (int)(t % CH);
    int s = src[e], d = dst[e];
    const float4 v = *(const float4*)(xin + s * FIN + c * 4);
    float* p = agg + d * FIN + c * 4;
    unsafeAtomicAdd(p + 0, v.x);
    unsafeAtomicAdd(p + 1, v.y);
    unsafeAtomicAdd(p + 2, v.z);
    unsafeAtomicAdd(p + 3, v.w);
}

// ---------- fused SAGE layer GEMM ----------
// out[n, 0:128] = act( [mean(agg,cnt) | xin] @ [Wl | Wr]^T + bl )
// 64 nodes / workgroup, 8 waves; wave w: row-tile = w>>1, 64-col group = w&1.
template<int FIN, bool RELU>
__launch_bounds__(256)
__global__ void sage_gemm_kernel(const float* __restrict__ xin,
                                 const float* __restrict__ agg,
                                 const float* __restrict__ cnt,
                                 const float* __restrict__ Wl,
                                 const float* __restrict__ Wr,
                                 const float* __restrict__ bl,
                                 float* __restrict__ out,
                                 int n_nodes) {
    constexpr int K     = 2 * FIN;   // concatenated K (128 or 256)
    constexpr int NODES = 64;
    constexpr int KC    = 64;        // B staging chunk along K

    __shared__ unsigned short Alds[NODES * K];  // <=32 KB (bf16 bits)
    __shared__ unsigned short Blds[128 * KC];   // 16 KB

    const int tid       = threadIdx.x;
    const int node_base = blockIdx.x * NODES;

    // Stage A = [mean | x] as bf16, zero-padded past n_nodes.
    for (int idx = tid; idx < NODES * K; idx += 256) {
        int r = idx / K, k = idx % K;
        int node = node_base + r;
        float v = 0.0f;
        if (node < n_nodes) {
            if (k < FIN) {
                float inv = 1.0f / fmaxf(cnt[node], 1.0f);
                v = agg[node * FIN + k] * inv;
            } else {
                v = xin[node * FIN + (k - FIN)];
            }
        }
        Alds[idx] = f2bf(v);
    }

    const int wave = tid >> 5;
    const int lane = tid & 31;
    const int rt   = wave >> 1;       // row tile 0..3
    const int cg   = wave & 1;        // column half 0..1
    const int mn   = lane & 15;       // M for A-frag, N for B-frag / C

    v8f acc[4] = {};

    for (int ch = 0; ch < K / KC; ++ch) {
        __syncthreads();  // A visible (iter 0) / previous chunk consumed
        // Stage B chunk: Blds[col][kloc] = W[col][ch*KC + kloc] (Wl|Wr concat)
        for (int idx = tid; idx < 128 * KC; idx += 256) {
            int n = idx / KC, kl = idx % KC;
            int kg = ch * KC + kl;
            float w = (kg < FIN) ? Wl[n * FIN + kg] : Wr[n * FIN + (kg - FIN)];
            Blds[idx] = f2bf(w);
        }
        __syncthreads();

        #pragma unroll
        for (int kk = 0; kk < KC; kk += 32) {
            union { v16bf v; unsigned short u[16]; } af;
            #pragma unroll
            for (int j = 0; j < 16; ++j)
                af.u[j] = Alds[(rt * 16 + mn) * K + ch * KC + kk + kpat(j, lane)];
            #pragma unroll
            for (int ct = 0; ct < 4; ++ct) {
                union { v16bf v; unsigned short u[16]; } bfv;
                #pragma unroll
                for (int j = 0; j < 16; ++j)
                    bfv.u[j] = Blds[(cg * 64 + ct * 16 + mn) * KC + kk + kpat(j, lane)];
                acc[ct] = __builtin_amdgcn_wmma_f32_16x16x32_bf16(
                    false, af.v, false, bfv.v, (short)0, acc[ct], false, false);
            }
        }
    }

    // C layout: VGPR v -> M = v + (lane>=16 ? 8 : 0); N = lane&15
    #pragma unroll
    for (int ct = 0; ct < 4; ++ct) {
        int col  = cg * 64 + ct * 16 + mn;
        float bias = bl[col];
        #pragma unroll
        for (int v = 0; v < 8; ++v) {
            int row  = rt * 16 + ((lane >> 4) << 3) + v;
            int node = node_base + row;
            if (node < n_nodes) {
                float val = acc[ct][v] + bias;
                if (RELU) val = fmaxf(val, 0.0f);
                out[node * 128 + col] = val;
            }
        }
    }
}

// ---------- global mean pool ----------
__global__ void pool_accum_kernel(const float* __restrict__ h,
                                  const int* __restrict__ batch,
                                  float* __restrict__ gsum,
                                  float* __restrict__ gcnt,
                                  int n_nodes) {
    long long t = (long long)blockIdx.x * blockDim.x + threadIdx.x;
    long long nwork = (long long)n_nodes * 32;
    if (t >= nwork) return;
    int node = (int)(t >> 5);
    int c    = (int)(t & 31);
    int g    = batch[node];
    const float4 v = *(const float4*)(h + node * 128 + c * 4);
    float* p = gsum + g * 128 + c * 4;
    unsafeAtomicAdd(p + 0, v.x);
    unsafeAtomicAdd(p + 1, v.y);
    unsafeAtomicAdd(p + 2, v.z);
    unsafeAtomicAdd(p + 3, v.w);
    if (c == 0) unsafeAtomicAdd(&gcnt[g], 1.0f);
}

__global__ void pool_final_kernel(const float* __restrict__ gsum,
                                  const float* __restrict__ gcnt,
                                  float* __restrict__ out) {
    int t = blockIdx.x * blockDim.x + threadIdx.x;
    if (t < 64 * 128) {
        int g = t / 128;
        out[t] = gsum[t] / fmaxf(gcnt[g], 1.0f);
    }
}

// ---------- host ----------
extern "C" void kernel_launch(void* const* d_in, const int* in_sizes, int n_in,
                              void* d_out, int out_size, void* d_ws, size_t ws_size,
                              hipStream_t stream) {
    const float* x     = (const float*)d_in[0];
    const int*   ei    = (const int*)d_in[1];
    const int*   batch = (const int*)d_in[2];
    const float* Wl1   = (const float*)d_in[3];
    const float* bl1   = (const float*)d_in[4];
    const float* Wr1   = (const float*)d_in[5];
    const float* Wl2   = (const float*)d_in[6];
    const float* bl2   = (const float*)d_in[7];
    const float* Wr2   = (const float*)d_in[8];
    const float* Wl3   = (const float*)d_in[9];
    const float* bl3   = (const float*)d_in[10];
    const float* Wr3   = (const float*)d_in[11];

    const int N = in_sizes[0] / 64;       // 50000 nodes (node_dim = 64)
    const int E = in_sizes[1] / 2;        // 800000 edges
    const int* src = ei;
    const int* dst = ei + E;

    // workspace layout (floats)
    float* ws = (float*)d_ws;
    size_t o = 0;
    float* cnt  = ws + o; o += ((size_t)N + 1023) & ~(size_t)1023;
    float* gsum = ws + o; o += 64 * 128;                 // followed directly by gcnt
    float* gcnt = ws + o; o += 1024;
    float* agg  = ws + o; o += (size_t)N * 128;
    float* hA   = ws + o; o += (size_t)N * 128;
    float* hB   = ws + o; o += (size_t)N * 128;
    (void)ws_size;

    const int TB = 256;
    const int ZB = 2048;  // grid-stride zero blocks
    auto cdiv = [](long long a, long long b) { return (int)((a + b - 1) / b); };

    // degree counts (shared across layers)
    zero_kernel<<<ZB, TB, 0, stream>>>(cnt, N);
    count_kernel<<<cdiv(E, TB), TB, 0, stream>>>(dst, cnt, E);

    // ---- layer 1 (Fin=64, relu) ----
    zero_kernel<<<ZB, TB, 0, stream>>>(agg, N * 64);
    scatter_kernel<64><<<cdiv((long long)E * 16, TB), TB, 0, stream>>>(x, src, dst, agg, E);
    sage_gemm_kernel<64, true><<<cdiv(N, 64), TB, 0, stream>>>(x, agg, cnt, Wl1, Wr1, bl1, hA, N);

    // ---- layer 2 (Fin=128, relu) ----
    zero_kernel<<<ZB, TB, 0, stream>>>(agg, N * 128);
    scatter_kernel<128><<<cdiv((long long)E * 32, TB), TB, 0, stream>>>(hA, src, dst, agg, E);
    sage_gemm_kernel<128, true><<<cdiv(N, 64), TB, 0, stream>>>(hA, agg, cnt, Wl2, Wr2, bl2, hB, N);

    // ---- layer 3 (Fin=128, no relu) ----
    zero_kernel<<<ZB, TB, 0, stream>>>(agg, N * 128);
    scatter_kernel<128><<<cdiv((long long)E * 32, TB), TB, 0, stream>>>(hB, src, dst, agg, E);
    sage_gemm_kernel<128, false><<<cdiv(N, 64), TB, 0, stream>>>(hB, agg, cnt, Wl3, Wr3, bl3, hA, N);

    // ---- global mean pool ----
    zero_kernel<<<16, TB, 0, stream>>>(gsum, 64 * 128 + 64);   // gsum + gcnt contiguous
    pool_accum_kernel<<<cdiv((long long)N * 32, TB), TB, 0, stream>>>(hA, batch, gsum, gcnt, N);
    pool_final_kernel<<<cdiv(64 * 128, TB), TB, 0, stream>>>(gsum, gcnt, (float*)d_out);
}
